// AdditiveAttention_9663676416057
// MI455X (gfx1250) — matrix operand order repaired and checked by reading, not compiled, not run on hardware
//
#include <hip/hip_runtime.h>
#include <hip/hip_bf16.h>

// ---------------------------------------------------------------------------
// Additive attention, fused for MI455X (gfx1250, wave32, WMMA bf16 path).
//   values:[256,196,2048] query:[256,512] W1:[2048,512] b1:[512]
//   W2:[512,512] b2:[512] v:[512] bv:[1]
//   out = (context:[256,2048], attn:[256,196]) concatenated.
// ---------------------------------------------------------------------------

typedef __attribute__((ext_vector_type(16))) __bf16 v16bf;
typedef __attribute__((ext_vector_type(2)))  __bf16 v2bf;
typedef __attribute__((ext_vector_type(8)))  float  v8f;
typedef __attribute__((ext_vector_type(4)))  float  vf4;

union FragBF {
    v16bf v;
    uint4 q[2];
};

__device__ __forceinline__ unsigned short f2bf(float f) {
    unsigned u = __float_as_uint(f);
    unsigned r = u + 0x7FFFu + ((u >> 16) & 1u);   // round-to-nearest-even
    return (unsigned short)(r >> 16);
}

// pack two f32 -> one dword of two bf16 (RNE). gfx1250 has v_cvt_pk_bf16_f32.
__device__ __forceinline__ unsigned pack2bf(float lo, float hi) {
#if __has_builtin(__builtin_amdgcn_cvt_pk_bf16_f32)
    v2bf r = __builtin_amdgcn_cvt_pk_bf16_f32(lo, hi);
    return __builtin_bit_cast(unsigned, r);
#else
    return (unsigned)f2bf(lo) | ((unsigned)f2bf(hi) << 16);
#endif
}

static constexpr int B_  = 256;
static constexpr int N_  = 196;
static constexpr int DV  = 2048;
static constexpr int A_  = 512;
static constexpr int M_  = B_ * N_;           // 50176, divisible by 32
static constexpr int KT  = DV / 32;           // 64 K-steps

// ---------------------------------------------------------------------------
// Kernel 1: repack W1 (fp32 [2048,512]) into WMMA-B bf16 fragments.
// Fragment (kt, ct) is 1 KB: lane l holds 16 bf16 at bytes [l*32, l*32+32):
//   element i -> K = kt*32 + (l<16 ? 0 : 8) + (i & 7) + (i<8 ? 0 : 16),  col = ct*16 + (l & 15)
// matching the ISA 16-bit B (32x16) layout.
// ---------------------------------------------------------------------------
__global__ __launch_bounds__(256) void packW1_kernel(
    const float* __restrict__ W1, unsigned short* __restrict__ packed)
{
    int e      = blockIdx.x * 256 + threadIdx.x;   // 0 .. 2048*512-1
    int frag   = e >> 9;
    int within = e & 511;
    int l      = within >> 4;
    int i      = within & 15;
    int kt     = frag >> 5;
    int ct     = frag & 31;
    int k      = kt * 32 + ((l & 16) ? 8 : 0) + (i & 7) + ((i & 8) ? 16 : 0);
    int col    = ct * 16 + (l & 15);
    packed[e]  = f2bf(W1[k * A_ + col]);
}

// ---------------------------------------------------------------------------
// Kernel 2: query_att = query @ W2 + b2   -> ws  [256, 512]  (tiny, fp32 VALU)
// ---------------------------------------------------------------------------
__global__ __launch_bounds__(256) void qatt_kernel(
    const float* __restrict__ query, const float* __restrict__ W2,
    const float* __restrict__ b2, float* __restrict__ qatt)
{
    int b = blockIdx.x, t = threadIdx.x;
    const float* qb = query + b * 512;
    float a0 = 0.f, a1 = 0.f;
#pragma unroll 8
    for (int k = 0; k < 512; ++k) {
        float q = qb[k];
        a0 = fmaf(q, W2[k * A_ + t],       a0);
        a1 = fmaf(q, W2[k * A_ + t + 256], a1);
    }
    qatt[b * A_ + t]       = a0 + b2[t];
    qatt[b * A_ + t + 256] = a1 + b2[t + 256];
}

// ---------------------------------------------------------------------------
// Kernel 3 (main): scores[m] = tanh(values[m,:]@W1 + b1 + qatt[b]) . v + bv
// Block: 256 threads = 8 waves, M-tile = 32 rows, full 512 output cols.
// Wave w -> m_sub = w>>2 (16 rows), col group = (w&3)*128 (8 16x16 tiles).
// Double-buffered LDS A-stage (fp32 -> bf16, row stride 80 B, conflict-free),
// one barrier per K-step, 8x v_wmma_f32_16x16x32_bf16 per step.
// ---------------------------------------------------------------------------
__global__ __launch_bounds__(256) void scores_kernel(
    const float* __restrict__ values, const unsigned short* __restrict__ w1p,
    const float* __restrict__ qatt,   const float* __restrict__ b1,
    const float* __restrict__ vvec,   const float* __restrict__ bv,
    float* __restrict__ scores)
{
    __shared__ __align__(16) unsigned char ldsA[2][32 * 80]; // 2 x (32 rows x 32 bf16, pad)
    __shared__ float ldsS[32];

    const int tid   = threadIdx.x;
    const int wave  = tid >> 5;
    const int lane  = tid & 31;
    const int m0    = blockIdx.x * 32;
    const int m_sub = wave >> 2;        // 0..1
    const int cg    = wave & 3;         // 0..3 (128 cols each)

    if (tid < 32) ldsS[tid] = 0.f;

    v8f acc[8] = {};                    // 8 C-tiles of 16x16 f32

    // cooperative A staging: thread -> (row, 4-float K group)
    const int    lrow = tid >> 3;       // 0..31
    const int    lk   = tid & 7;        // 0..7
    const float* asrc = values + (size_t)(m0 + lrow) * DV + (size_t)lk * 4;
    const unsigned soff = (unsigned)(lrow * 80 + lk * 8);

    // A-fragment read pattern (ISA 16-bit A 16x32 layout):
    //  lane<16: M=lane,    K in {0..7, 16..23}; lane>=16: M=lane-16, K in {8..15, 24..31}
    const int      hi16  = (lane >> 4) & 1;
    const unsigned aoff0 = (unsigned)(((m_sub << 4) + (lane & 15)) * 80 + hi16 * 16);
    const unsigned lbyte = (unsigned)lane * 32u;

    // prologue: stage K-chunk 0 into buffer 0
    {
        const vf4 f = __builtin_nontemporal_load((const vf4*)asrc);
        uint2 pk;
        pk.x = pack2bf(f.x, f.y);
        pk.y = pack2bf(f.z, f.w);
        *(uint2*)(ldsA[0] + soff) = pk;
    }

    int cur = 0;
#pragma unroll 1
    for (int kt = 0; kt < KT - 1; ++kt) {
        __syncthreads();
        // prefetch next A chunk (non-temporal: protect L2 for W1/qatt)
        const vf4 fn = __builtin_nontemporal_load(
            (const vf4*)(asrc + (size_t)(kt + 1) * 32));

        FragBF Af;
        Af.q[0] = *(const uint4*)(ldsA[cur] + aoff0);
        Af.q[1] = *(const uint4*)(ldsA[cur] + aoff0 + 32);

        const unsigned short* bbase = w1p + (size_t)(kt * 32 + cg * 8) * 512;
#pragma unroll
        for (int t = 0; t < 8; ++t) {
            FragBF Bf;
            const unsigned char* bp = (const unsigned char*)(bbase + (size_t)t * 512);
            Bf.q[0] = *(const uint4*)(bp + lbyte);
            Bf.q[1] = *(const uint4*)(bp + lbyte + 16);
            acc[t] = __builtin_amdgcn_wmma_f32_16x16x32_bf16(
                false, Af.v, false, Bf.v, (short)0, acc[t], false, false);
        }

        // convert + store next chunk into the other buffer (read next iter)
        uint2 pk;
        pk.x = pack2bf(fn.x, fn.y);
        pk.y = pack2bf(fn.z, fn.w);
        *(uint2*)(ldsA[cur ^ 1] + soff) = pk;
        cur ^= 1;
    }
    // final K-step (no prefetch / no store)
    {
        __syncthreads();
        FragBF Af;
        Af.q[0] = *(const uint4*)(ldsA[cur] + aoff0);
        Af.q[1] = *(const uint4*)(ldsA[cur] + aoff0 + 32);
        const unsigned short* bbase = w1p + (size_t)((KT - 1) * 32 + cg * 8) * 512;
#pragma unroll
        for (int t = 0; t < 8; ++t) {
            FragBF Bf;
            const unsigned char* bp = (const unsigned char*)(bbase + (size_t)t * 512);
            Bf.q[0] = *(const uint4*)(bp + lbyte);
            Bf.q[1] = *(const uint4*)(bp + lbyte + 16);
            acc[t] = __builtin_amdgcn_wmma_f32_16x16x32_bf16(
                false, Af.v, false, Bf.v, (short)0, acc[t], false, false);
        }
    }

    // ---- fused epilogue: +b1 +qatt -> tanh -> dot with v -> scores --------
    // C layout: vgpr r, lane<16 -> M=r, N=lane; lane>=16 -> M=r+8, N=lane-16
    int brow[8];
#pragma unroll
    for (int r = 0; r < 8; ++r)
        brow[r] = (m0 + (m_sub << 4) + r + (hi16 ? 8 : 0)) / N_;

    float part[8] = {};
#pragma unroll
    for (int t = 0; t < 8; ++t) {
        const int   col = (cg << 7) + (t << 4) + (lane & 15);
        const float b1v = b1[col];
        const float vv  = vvec[col];
#pragma unroll
        for (int r = 0; r < 8; ++r) {
            float x = acc[t][r] + b1v + qatt[brow[r] * A_ + col];
            part[r] = fmaf(tanhf(x), vv, part[r]);
        }
    }
#pragma unroll
    for (int r = 0; r < 8; ++r) {
        float p = part[r];
        for (int m = 1; m < 16; m <<= 1) p += __shfl_xor(p, m, 32);
        if ((lane & 15) == 0)
            atomicAdd(&ldsS[(m_sub << 4) + r + (hi16 ? 8 : 0)], p);
    }
    __syncthreads();
    if (tid < 32) scores[m0 + tid] = ldsS[tid] + bv[0];
}

// ---------------------------------------------------------------------------
// Kernel 4: softmax over N=196 per batch
// ---------------------------------------------------------------------------
__global__ __launch_bounds__(256) void softmax_kernel(
    const float* __restrict__ scores, float* __restrict__ attn)
{
    __shared__ float red[256];
    int b = blockIdx.x, t = threadIdx.x;
    float s = (t < N_) ? scores[b * N_ + t] : -3.4e38f;
    red[t] = s; __syncthreads();
    for (int o = 128; o > 0; o >>= 1) {
        if (t < o) red[t] = fmaxf(red[t], red[t + o]);
        __syncthreads();
    }
    float mx = red[0]; __syncthreads();
    float e = (t < N_) ? __expf(s - mx) : 0.f;
    red[t] = e; __syncthreads();
    for (int o = 128; o > 0; o >>= 1) {
        if (t < o) red[t] += red[t + o];
        __syncthreads();
    }
    float inv = 1.f / red[0];
    if (t < N_) attn[b * N_ + t] = e * inv;
}

// ---------------------------------------------------------------------------
// Kernel 5: context[b,:] = sum_n values[b,n,:] * attn[b,n]   (memory-bound,
// non-temporal streaming of values; NT store of context)
// ---------------------------------------------------------------------------
__global__ __launch_bounds__(256) void context_kernel(
    const float* __restrict__ values, const float* __restrict__ attn,
    float* __restrict__ ctx)
{
    int b    = blockIdx.x >> 1;
    int col  = (blockIdx.x & 1) * 1024 + threadIdx.x * 4;
    vf4 acc = {0.f, 0.f, 0.f, 0.f};
    const float* vb = values + (size_t)b * N_ * DV + col;
#pragma unroll 4
    for (int n = 0; n < N_; ++n) {
        float a = attn[b * N_ + n];
        vf4   x = __builtin_nontemporal_load((const vf4*)(vb + (size_t)n * DV));
        acc.x = fmaf(a, x.x, acc.x);
        acc.y = fmaf(a, x.y, acc.y);
        acc.z = fmaf(a, x.z, acc.z);
        acc.w = fmaf(a, x.w, acc.w);
    }
    __builtin_nontemporal_store(acc, (vf4*)(ctx + (size_t)b * DV + col));
}

// ---------------------------------------------------------------------------
extern "C" void kernel_launch(void* const* d_in, const int* in_sizes, int n_in,
                              void* d_out, int out_size, void* d_ws, size_t ws_size,
                              hipStream_t stream)
{
    const float* values = (const float*)d_in[0];
    const float* query  = (const float*)d_in[1];
    const float* W1     = (const float*)d_in[2];
    const float* b1     = (const float*)d_in[3];
    const float* W2     = (const float*)d_in[4];
    const float* b2     = (const float*)d_in[5];
    const float* v      = (const float*)d_in[6];
    const float* bv     = (const float*)d_in[7];

    float* out_ctx  = (float*)d_out;                 // [256, 2048]
    float* out_attn = out_ctx + (size_t)B_ * DV;     // [256, 196]

    // workspace layout (~2.8 MB total)
    char* ws = (char*)d_ws;
    unsigned short* w1p   = (unsigned short*)ws;                         // 2 MB
    float*          qatt  = (float*)(ws + (size_t)DV * A_ * 2);          // 512 KB
    float*          score = (float*)((char*)qatt + (size_t)B_ * A_ * 4); // 196 KB

    packW1_kernel <<<(DV * A_) / 256, 256, 0, stream>>>(W1, w1p);
    qatt_kernel   <<<B_,             256, 0, stream>>>(query, W2, b2, qatt);
    scores_kernel <<<M_ / 32,        256, 0, stream>>>(values, w1p, qatt, b1, v, bv, score);
    softmax_kernel<<<B_,             256, 0, stream>>>(score, out_attn);
    context_kernel<<<B_ * 2,         256, 0, stream>>>(values, out_attn, out_ctx);
}
